// PointNetSetAbstractionMsgDynamic_38268158607592
// MI455X (gfx1250) — matrix operand order repaired and checked by reading, not compiled, not run on hardware
//
#include <hip/hip_runtime.h>
#include <hip/hip_bf16.h>

#define N_PTS   65536
#define P_PIECES 16
#define NP_PIECE 4096
#define SP_PIECE 1024
#define S_TOT   16384
#define IN_CH   9

typedef __attribute__((ext_vector_type(16))) _Float16 v16h;
typedef __attribute__((ext_vector_type(8)))  float    v8f;

// ---------------------------------------------------------------------------
// 1) Farthest point sampling: one block per piece. Coords in LDS, per-thread
//    register distance cache (16 pts/thread), LDS tree argmax.
// ---------------------------------------------------------------------------
__global__ __launch_bounds__(256) void fps_kernel(const float* __restrict__ xyz,
                                                  int* __restrict__ centroids) {
  __shared__ float sx[NP_PIECE], sy[NP_PIECE], sz[NP_PIECE];
  __shared__ float rv[256];
  __shared__ int   ri[256];
  __shared__ int   s_cur;
  const int p = blockIdx.x;
  const int t = threadIdx.x;
  const int base = p * NP_PIECE;
  for (int i = t; i < NP_PIECE; i += 256) {
    sx[i] = xyz[base + i];
    sy[i] = xyz[N_PTS + base + i];
    sz[i] = xyz[2 * N_PTS + base + i];
  }
  float dist[16];
#pragma unroll
  for (int i = 0; i < 16; ++i) dist[i] = 3.4e38f;
  if (t == 0) { s_cur = 0; centroids[p * SP_PIECE] = base; }
  __syncthreads();
  int cur = 0;
  for (int it = 1; it < SP_PIECE; ++it) {
    const float cx = sx[cur], cy = sy[cur], cz = sz[cur];
    float best = -1.0f; int bi = 0;
#pragma unroll
    for (int i = 0; i < 16; ++i) {
      const int pidx = i * 256 + t;
      const float dx = sx[pidx] - cx, dy = sy[pidx] - cy, dz = sz[pidx] - cz;
      const float d = dx * dx + dy * dy + dz * dz;
      const float dd = fminf(dist[i], d);
      dist[i] = dd;
      if (dd > best) { best = dd; bi = pidx; }
    }
    rv[t] = best; ri[t] = bi;
    __syncthreads();
    for (int stride = 128; stride > 0; stride >>= 1) {
      if (t < stride) {
        const float o = rv[t + stride];
        if (o > rv[t] || (o == rv[t] && ri[t + stride] < ri[t])) {
          rv[t] = o; ri[t] = ri[t + stride];
        }
      }
      __syncthreads();
    }
    if (t == 0) { s_cur = ri[0]; centroids[p * SP_PIECE + it] = base + ri[0]; }
    __syncthreads();
    cur = s_cur;
  }
}

// ---------------------------------------------------------------------------
// 2) Emit new_xyz (3*S) and new_pid (S) into d_out.
// ---------------------------------------------------------------------------
__global__ void write_newxyz_kernel(const float* __restrict__ xyz,
                                    const int* __restrict__ piece_id,
                                    const int* __restrict__ centroids,
                                    float* __restrict__ out) {
  const int s = blockIdx.x * blockDim.x + threadIdx.x;
  if (s >= S_TOT) return;
  const int c = centroids[s];
  out[s]             = xyz[c];
  out[S_TOT + s]     = xyz[N_PTS + c];
  out[2 * S_TOT + s] = xyz[2 * N_PTS + c];
  out[3 * S_TOT + s] = (float)piece_id[c];
}

// ---------------------------------------------------------------------------
// 3) kNN (top-64, sorted ascending; top-32 = prefix). One thread per centroid,
//    per-thread sorted list in LDS laid out [entry*64 + tid] (conflict-free).
// ---------------------------------------------------------------------------
__global__ __launch_bounds__(64) void knn_kernel(const float* __restrict__ xyz,
                                                 const int* __restrict__ centroids,
                                                 int* __restrict__ gidx) {
  __shared__ float cd[64 * 64];
  __shared__ int   ci[64 * 64];
  const int t = threadIdx.x;
  const int p = blockIdx.x;
  const int g = blockIdx.y;
  const int s = p * SP_PIECE + g * 64 + t;
  const int base = p * NP_PIECE;
  const int c = centroids[s];
  const float cx = xyz[c], cy = xyz[N_PTS + c], cz = xyz[2 * N_PTS + c];
#pragma unroll 4
  for (int j = 0; j < 64; ++j) cd[j * 64 + t] = 3.4e38f;
  for (int jj = 0; jj < NP_PIECE; ++jj) {
    const float px = xyz[base + jj];
    const float py = xyz[N_PTS + base + jj];
    const float pz = xyz[2 * N_PTS + base + jj];
    const float dx = px - cx, dy = py - cy, dz = pz - cz;
    const float d2 = dx * dx + dy * dy + dz * dz;
    if (d2 < cd[63 * 64 + t]) {
      int pos = 63;
      while (pos > 0 && cd[(pos - 1) * 64 + t] > d2) {
        cd[pos * 64 + t] = cd[(pos - 1) * 64 + t];
        ci[pos * 64 + t] = ci[(pos - 1) * 64 + t];
        --pos;
      }
      cd[pos * 64 + t] = d2;
      ci[pos * 64 + t] = jj;
    }
  }
  for (int j = 0; j < 64; ++j) gidx[(size_t)s * 64 + j] = base + ci[j * 64 + t];
}

// ---------------------------------------------------------------------------
// 4) Gather layer-0 inputs: row = [feat(9), relative xyz(3), pad->32] as f16.
// ---------------------------------------------------------------------------
__global__ void gather_kernel(const float* __restrict__ xyz,
                              const float* __restrict__ points,
                              const int* __restrict__ centroids,
                              const int* __restrict__ gidx,
                              _Float16* __restrict__ X0, int K, int Mtot) {
  const int r = blockIdx.x * blockDim.x + threadIdx.x;
  if (r >= Mtot) return;
  const int s = r / K;
  const int k = r - s * K;
  const int g = gidx[(size_t)s * 64 + k];
  const int c = centroids[s];
  alignas(16) _Float16 row[32];
#pragma unroll
  for (int j = 0; j < IN_CH; ++j) row[j] = (_Float16)points[(size_t)j * N_PTS + g];
#pragma unroll
  for (int d = 0; d < 3; ++d)
    row[IN_CH + d] = (_Float16)(xyz[(size_t)d * N_PTS + g] - xyz[(size_t)d * N_PTS + c]);
#pragma unroll
  for (int j = 12; j < 32; ++j) row[j] = (_Float16)0.0f;
  uint4* dst = (uint4*)(X0 + (size_t)r * 32);
  const uint4* src = (const uint4*)row;
#pragma unroll
  for (int q = 0; q < 4; ++q) dst[q] = src[q];
}

// ---------------------------------------------------------------------------
// 5) Pack weights into the WMMA B-lane layout as f16 (zero-padded K):
//    element (ntile, kk, lane, j) = W[ntile*16+(lane&15)][kk*32 + 8*(lane>>4) + (j<8 ? j : 8+j)]
// ---------------------------------------------------------------------------
__global__ void prep_weights_kernel(const float* __restrict__ W,
                                    _Float16* __restrict__ Bp,
                                    int Cin, int CinPad, int Cout) {
  const int ksteps = CinPad >> 5;
  const int total = (Cout >> 4) * ksteps * 32;
  const int t = blockIdx.x * blockDim.x + threadIdx.x;
  if (t >= total) return;
  const int lane = t & 31;
  const int kk = (t >> 5) % ksteps;
  const int ntile = t / (32 * ksteps);
  const int nrow = ntile * 16 + (lane & 15);
  alignas(16) _Float16 row[16];
#pragma unroll
  for (int j = 0; j < 16; ++j) {
    const int kb = kk * 32 + 8 * (lane >> 4) + (j < 8 ? j : 8 + j);
    row[j] = (kb < Cin) ? (_Float16)W[(size_t)nrow * Cin + kb]
                        : (_Float16)0.0f;
  }
  uint4* dst = (uint4*)(Bp + (size_t)t * 16);
  const uint4* src = (const uint4*)row;
  dst[0] = src[0];
  dst[1] = src[1];
}

// ---------------------------------------------------------------------------
// 6) WMMA GEMM: Y[M,Cout] = X[M,CinPad] * W^T + b.  One wave computes FOUR
//    consecutive 16x16 M-tiles sharing one B-tile; KSTEPS is compile-time so
//    the whole multiply is a straight-line run of 4*KSTEPS v_wmma ops fed by
//    b128 loads.  Writes pre-BN f16 output + per-channel sum/sumsq.
// ---------------------------------------------------------------------------
template <int KSTEPS>
__global__ __launch_bounds__(256) void mlp_layer_kernel(
    const _Float16* __restrict__ Xin, const _Float16* __restrict__ Bp,
    const float* __restrict__ bias,
    _Float16* __restrict__ Xout,
    float* __restrict__ sums, float* __restrict__ sumsq,
    int Cout) {
  constexpr int CinPad = KSTEPS * 32;
  __shared__ float bsum[16], bsq[16];
  const int lane = threadIdx.x & 31;
  const int wave = threadIdx.x >> 5;
  if (threadIdx.x < 16) { bsum[threadIdx.x] = 0.0f; bsq[threadIdx.x] = 0.0f; }
  __syncthreads();

  const int mt0 = (blockIdx.x * 8 + wave) * 4;   // first of 4 M-tiles
  const int ntile = blockIdx.y;
  const int hi = lane >> 4;
  const int lm = lane & 15;
  const int nch = ntile * 16 + lm;

  v8f acc[4];
  const float bv = bias[nch];
#pragma unroll
  for (int tt = 0; tt < 4; ++tt)
#pragma unroll
    for (int r2 = 0; r2 < 8; ++r2) acc[tt][r2] = bv;

  // A lane owns row M = mtile*16+lm, K chunks {8hi..8hi+7} and {16+8hi..+23}
  const _Float16* pa = Xin + (size_t)(mt0 * 16 + lm) * CinPad + 8 * hi;
  // Packed B: contiguous 16 halves per (ntile, kk, lane)
  const _Float16* pb = Bp + ((size_t)(ntile * KSTEPS) * 32 + lane) * 16;

#pragma unroll
  for (int kk = 0; kk < KSTEPS; ++kk) {
    union { uint4 u[2]; v16h v; } bu;
    bu.u[0] = *(const uint4*)(pb + kk * 512);
    bu.u[1] = *(const uint4*)(pb + kk * 512 + 8);
#pragma unroll
    for (int tt = 0; tt < 4; ++tt) {
      const _Float16* pr = pa + (size_t)tt * 16 * CinPad + kk * 32;
      union { uint4 u[2]; v16h v; } au;
      au.u[0] = *(const uint4*)(pr);
      au.u[1] = *(const uint4*)(pr + 16);
      acc[tt] = __builtin_amdgcn_wmma_f32_16x16x32_f16(false, au.v, false, bu.v,
                                                       (short)0, acc[tt], false, false);
    }
  }

  float s1 = 0.0f, s2 = 0.0f;
#pragma unroll
  for (int tt = 0; tt < 4; ++tt) {
#pragma unroll
    for (int r2 = 0; r2 < 8; ++r2) {
      const int m = (mt0 + tt) * 16 + r2 + 8 * hi;
      const float v = acc[tt][r2];
      Xout[(size_t)m * Cout + nch] = (_Float16)v;
      s1 += v;
      s2 += v * v;
    }
  }
  atomicAdd(&bsum[lm], s1);
  atomicAdd(&bsq[lm], s2);
  __syncthreads();
  if (threadIdx.x < 16) {
    atomicAdd(&sums[ntile * 16 + threadIdx.x], bsum[threadIdx.x]);
    atomicAdd(&sumsq[ntile * 16 + threadIdx.x], bsq[threadIdx.x]);
  }
}

// ---------------------------------------------------------------------------
// 7) Fold batch stats + (gamma, beta) into per-channel scale/shift.
// ---------------------------------------------------------------------------
__global__ void bn_finalize_kernel(const float* __restrict__ sums,
                                   const float* __restrict__ sumsq,
                                   const float* __restrict__ g,
                                   const float* __restrict__ be,
                                   float* __restrict__ scale,
                                   float* __restrict__ shift,
                                   int C, float invM) {
  const int c = blockIdx.x * blockDim.x + threadIdx.x;
  if (c >= C) return;
  const float mean = sums[c] * invM;
  const float var = sumsq[c] * invM - mean * mean;
  const float sc = g[c] * rsqrtf(var + 1e-5f);
  scale[c] = sc;
  shift[c] = be[c] - mean * sc;
}

__global__ void zero_f32_kernel(float* __restrict__ p, int n) {
  const int i = blockIdx.x * blockDim.x + threadIdx.x;
  if (i < n) p[i] = 0.0f;
}

// ---------------------------------------------------------------------------
// 8) In-place BN+ReLU over a pre-BN f16 activation buffer (8 halves/thread).
// ---------------------------------------------------------------------------
__global__ void bn_apply_kernel(_Float16* __restrict__ X,
                                const float* __restrict__ scale,
                                const float* __restrict__ shift,
                                int Cout, long total8) {
  const long t = (long)blockIdx.x * blockDim.x + threadIdx.x;
  if (t >= total8) return;
  _Float16* p = X + t * 8;
  const int c0 = (int)((t * 8) % Cout);
  union { uint4 u; _Float16 h[8]; } d;
  d.u = *(const uint4*)p;
#pragma unroll
  for (int i = 0; i < 8; ++i)
    d.h[i] = (_Float16)fmaxf(fmaf((float)d.h[i], scale[c0 + i], shift[c0 + i]), 0.0f);
  *(uint4*)p = d.u;
}

// ---------------------------------------------------------------------------
// 9) Apply BN+ReLU of the last layer and max-pool over K; write to d_out.
// ---------------------------------------------------------------------------
__global__ void maxpool_kernel(const _Float16* __restrict__ X,
                               const float* __restrict__ scale,
                               const float* __restrict__ shift,
                               float* __restrict__ out, int K, int Cout) {
  const int t = blockIdx.x * blockDim.x + threadIdx.x;
  if (t >= S_TOT * Cout) return;
  const int s = t / Cout;
  const int c = t - s * Cout;
  const float sc = scale[c], sh = shift[c];
  const _Float16* p = X + (size_t)s * K * Cout + c;
  float m = 0.0f;  // post-ReLU values are >= 0
  for (int k = 0; k < K; ++k) {
    const float v = fmaxf(fmaf((float)p[(size_t)k * Cout], sc, sh), 0.0f);
    m = fmaxf(m, v);
  }
  out[(size_t)c * S_TOT + s] = m;
}

// ---------------------------------------------------------------------------
extern "C" void kernel_launch(void* const* d_in, const int* in_sizes, int n_in,
                              void* d_out, int out_size, void* d_ws, size_t ws_size,
                              hipStream_t stream) {
  (void)in_sizes; (void)n_in; (void)out_size; (void)ws_size;
  const float* xyz      = (const float*)d_in[0];
  const int*   piece_id = (const int*)d_in[1];
  const float* points   = (const float*)d_in[2];

  char* ws = (char*)d_ws;
  int* centroids = (int*)ws;                           // 64 KB
  int* gidx      = (int*)(ws + (1 << 16));             // 4 MB
  float* sums    = (float*)(ws + 0x500000);            // 128 f32
  float* sumsq   = sums + 128;
  float* scaleB  = sums + 256;
  float* shiftB  = sums + 384;
  _Float16* Bp   = (_Float16*)(ws + 0x600000);         // packed weights, <=16 KB
  _Float16* bufA = (_Float16*)(ws + ((size_t)32  << 20));  // up to 128 MB
  _Float16* bufB = (_Float16*)(ws + ((size_t)160 << 20));  // up to 256 MB

  fps_kernel<<<P_PIECES, 256, 0, stream>>>(xyz, centroids);
  write_newxyz_kernel<<<(S_TOT + 255) / 256, 256, 0, stream>>>(xyz, piece_id,
                                                               centroids, (float*)d_out);
  knn_kernel<<<dim3(P_PIECES, SP_PIECE / 64), 64, 0, stream>>>(xyz, centroids, gidx);

  for (int b = 0; b < 2; ++b) {
    const int K = b ? 64 : 32;
    const int M = S_TOT * K;
    const int pbase = 3 + b * 12;
    const int Cin[3]    = {12, b ? 64 : 32, b ? 64 : 32};
    const int CinPad[3] = {32, b ? 64 : 32, b ? 64 : 32};
    const int Cout[3]   = {b ? 64 : 32, b ? 64 : 32, b ? 128 : 64};
    _Float16* act[4] = {bufA, bufB, bufA, bufB};

    gather_kernel<<<(M + 255) / 256, 256, 0, stream>>>(xyz, points, centroids,
                                                       gidx, bufA, K, M);
    for (int l = 0; l < 3; ++l) {
      const float* W  = (const float*)d_in[pbase + 4 * l + 0];
      const float* bb = (const float*)d_in[pbase + 4 * l + 1];
      const float* gg = (const float*)d_in[pbase + 4 * l + 2];
      const float* be = (const float*)d_in[pbase + 4 * l + 3];
      const int nB = (Cout[l] / 16) * (CinPad[l] / 32) * 32;
      prep_weights_kernel<<<(nB + 63) / 64, 64, 0, stream>>>(W, Bp, Cin[l],
                                                             CinPad[l], Cout[l]);
      zero_f32_kernel<<<1, 256, 0, stream>>>(sums, 256);
      dim3 grid(M / 512, Cout[l] / 16);   // 4 M-tiles per wave, 8 waves/block
      if (CinPad[l] == 32) {
        mlp_layer_kernel<1><<<grid, 256, 0, stream>>>(act[l], Bp, bb, act[l + 1],
                                                      sums, sumsq, Cout[l]);
      } else {
        mlp_layer_kernel<2><<<grid, 256, 0, stream>>>(act[l], Bp, bb, act[l + 1],
                                                      sums, sumsq, Cout[l]);
      }
      bn_finalize_kernel<<<1, 128, 0, stream>>>(sums, sumsq, gg, be, scaleB,
                                                shiftB, Cout[l], 1.0f / (float)M);
      if (l < 2) {
        const long total8 = (long)M * Cout[l] / 8;
        bn_apply_kernel<<<(int)((total8 + 255) / 256), 256, 0, stream>>>(
            act[l + 1], scaleB, shiftB, Cout[l], total8);
      }
    }
    maxpool_kernel<<<(S_TOT * Cout[2] + 255) / 256, 256, 0, stream>>>(
        act[3], scaleB, shiftB,
        (float*)d_out + 4 * S_TOT + (size_t)(b ? 64 : 0) * S_TOT, K, Cout[2]);
  }
}
